// EGAAltAttention_48816598286893
// MI455X (gfx1250) — matrix operand order repaired and checked by reading, not compiled, not run on hardware
//
#include <hip/hip_runtime.h>
#include <hip/hip_bf16.h>
#include <cstdint>

typedef __attribute__((ext_vector_type(16))) __bf16 v16bf;
typedef __attribute__((ext_vector_type(8)))  __bf16 v8bf;
typedef __attribute__((ext_vector_type(8)))  float  v8f;

namespace {
constexpr int kB   = 4;
constexpr int kN   = 1024;
constexpr int kDIM = 1024;
constexpr int kH   = 16;
constexpr int kD   = 64;
constexpr float kScale    = 0.125f;  // D^-0.5
constexpr float kSuppress = 0.1f;
}

// ---- helpers -------------------------------------------------------------

__device__ __forceinline__ __bf16 f2bf(float f) {
  uint32_t u = __builtin_bit_cast(uint32_t, f);
  uint32_t r = (u + 0x7FFFu + ((u >> 16) & 1u)) >> 16;  // RNE
  uint16_t s = (uint16_t)r;
  return __builtin_bit_cast(__bf16, s);
}

__device__ __forceinline__ v8f wmma_bf16(v16bf a, v16bf b, v8f c) {
  return __builtin_amdgcn_wmma_f32_16x16x32_bf16(false, a, false, b,
                                                 (short)0, c, false, false);
}

// A fragment: 16x32 (MxK) bf16. Caller passes &A[row = lane&15][k0].
// Per ISA: lanes 0-15 hold K = 0..7 & 16..23; lanes 16-31 hold K = 8..15 & 24..31.
__device__ __forceinline__ v16bf load_a_frag(const __bf16* rowk, int lane) {
  const int base = (lane & 16) ? 8 : 0;
  v8bf lo = *reinterpret_cast<const v8bf*>(rowk + base);
  v8bf hv = *reinterpret_cast<const v8bf*>(rowk + base + 16);
  v16bf a;
#pragma unroll
  for (int i = 0; i < 8; ++i) { a[i] = lo[i]; a[8 + i] = hv[i]; }
  return a;
}

// B fragment: 32x16 (KxN) bf16, column n = lane&15. Caller passes a pointer to
// that column's K-run (contiguous along K). Lanes 0-15: K=0..7 & 16..23, lanes 16-31: K=8..15 & 24..31.
__device__ __forceinline__ v16bf load_b_frag(const __bf16* colk, int lane) {
  const int base = (lane & 16) ? 8 : 0;
  v8bf lo = *reinterpret_cast<const v8bf*>(colk + base);
  v8bf hv = *reinterpret_cast<const v8bf*>(colk + base + 16);
  v16bf b;
#pragma unroll
  for (int i = 0; i < 8; ++i) { b[i] = lo[i]; b[8 + i] = hv[i]; }
  return b;
}

// ---- f32 -> bf16 conversion ---------------------------------------------

__global__ void cvt_bf16_kernel(const float* __restrict__ src,
                                __bf16* __restrict__ dst, int n) {
  int i = blockIdx.x * blockDim.x + threadIdx.x;
  if (i < n) dst[i] = f2bf(src[i]);
}

// ---- QKV projection: X[4096,1024] @ W^T + b ------------------------------
// Each wave computes a 16x64 strip (4 accumulators share one A fragment).
// All 10 b128 loads of a K-chunk are issued before its 4 WMMAs so the
// scheduler can retire them with partial s_wait_loadcnt under the MMAs.
// q,k stored [B,H,N,D]; v stored transposed [B,H,D,N] for the attn*V WMMA.

__global__ __launch_bounds__(128) void qkv_proj_kernel(
    const __bf16* __restrict__ xbf,
    const __bf16* __restrict__ Wqb, const __bf16* __restrict__ Wkb,
    const __bf16* __restrict__ Wvb,
    const float* __restrict__ bq, const float* __restrict__ bk,
    const float* __restrict__ bv,
    __bf16* __restrict__ qo, __bf16* __restrict__ ko,
    __bf16* __restrict__ vto) {
  const int lane  = threadIdx.x & 31;
  const int wv    = threadIdx.x >> 5;
  const int mt    = blockIdx.y * 4 + wv;   // 256 row tiles (16 rows each)
  const int cb    = blockIdx.x;            // 16 col blocks (64 cols each) == head
  const int which = blockIdx.z;            // 0=q 1=k 2=v
  const __bf16* W    = (which == 0) ? Wqb : (which == 1) ? Wkb : Wvb;
  const float*  bias = (which == 0) ? bq  : (which == 1) ? bk  : bv;
  const int row0 = mt * 16, col0 = cb * 64;
  const int nc = lane & 15, hi = (lane >> 4) & 1;
  const __bf16* arow  = xbf + (size_t)(row0 + nc) * kDIM;
  const __bf16* brow0 = W   + (size_t)(col0 + nc) * kDIM;
  v8f acc[4] = {{}, {}, {}, {}};
  for (int k = 0; k < kDIM; k += 32) {
    v16bf a  = load_a_frag(arow + k, lane);
    v16bf b0 = load_b_frag(brow0 + (size_t)0 * 16 * kDIM + k, lane);
    v16bf b1 = load_b_frag(brow0 + (size_t)1 * 16 * kDIM + k, lane);
    v16bf b2 = load_b_frag(brow0 + (size_t)2 * 16 * kDIM + k, lane);
    v16bf b3 = load_b_frag(brow0 + (size_t)3 * 16 * kDIM + k, lane);
    acc[0] = wmma_bf16(a, b0, acc[0]);
    acc[1] = wmma_bf16(a, b1, acc[1]);
    acc[2] = wmma_bf16(a, b2, acc[2]);
    acc[3] = wmma_bf16(a, b3, acc[3]);
  }
  const int hh = cb;                        // 64-col strip == one head
#pragma unroll
  for (int t = 0; t < 4; ++t) {
    const int dd  = t * 16 + nc;
    const float bsc = bias[col0 + dd];
#pragma unroll
    for (int r = 0; r < 8; ++r) {
      const int m  = row0 + r + 8 * hi;     // token row in [0, B*N)
      const int bb = m >> 10;
      const int n  = m & (kN - 1);
      const float val = acc[t][r] + bsc;
      if (which == 2) {
        vto[((size_t)(bb * kH + hh) * kD + dd) * kN + n] = f2bf(val);
      } else {
        __bf16* o = (which == 0) ? qo : ko;
        o[((size_t)(bb * kH + hh) * kN + n) * kD + dd] = f2bf(val);
      }
    }
  }
}

// ---- Flash attention: one wave per 16 query rows, online softmax ---------

__global__ __launch_bounds__(128) void flash_attn_kernel(
    const __bf16* __restrict__ qbf, const __bf16* __restrict__ kbf,
    const __bf16* __restrict__ vtbf, const float* __restrict__ alibi,
    const unsigned char* __restrict__ pmask, __bf16* __restrict__ Abf) {
  __shared__ float plds[4][16 * 32];       // per-wave 16x32 P tile (C->A relayout)
  const int lane = threadIdx.x & 31;
  const int wv   = threadIdx.x >> 5;
  const int q0   = (blockIdx.x * 4 + wv) * 16;
  const int h = blockIdx.y, b = blockIdx.z;
  const int nc = lane & 15, hi = (lane >> 4) & 1;
  const __bf16* qp  = qbf  + (size_t)(b * kH + h) * kN * kD;
  const __bf16* kp  = kbf  + (size_t)(b * kH + h) * kN * kD;
  const __bf16* vtp = vtbf + (size_t)(b * kH + h) * kD * kN;
  const v16bf qa0 = load_a_frag(qp + (size_t)(q0 + nc) * kD, lane);
  const v16bf qa1 = load_a_frag(qp + (size_t)(q0 + nc) * kD + 32, lane);
  float mrow[8], lrow[8];
  v8f o[4];
  v8f vzero = {};
#pragma unroll
  for (int r = 0; r < 8; ++r) { mrow[r] = -1e30f; lrow[r] = 0.f; }
#pragma unroll
  for (int d = 0; d < 4; ++d) o[d] = vzero;
  float* myp = plds[wv];

  for (int j = 0; j < kN; j += 32) {
    // S(16x32) = Q(16x64) x K^T(64x32): issue all 4 K fragments, then 4 WMMAs
    const __bf16* krow0 = kp + (size_t)(j + nc) * kD;
    const __bf16* krow1 = kp + (size_t)(j + 16 + nc) * kD;
    v16bf kb00 = load_b_frag(krow0, lane);
    v16bf kb01 = load_b_frag(krow0 + 32, lane);
    v16bf kb10 = load_b_frag(krow1, lane);
    v16bf kb11 = load_b_frag(krow1 + 32, lane);
    v8f s[2];
    s[0] = wmma_bf16(qa1, kb01, wmma_bf16(qa0, kb00, vzero));
    s[1] = wmma_bf16(qa1, kb11, wmma_bf16(qa0, kb10, vzero));
    // scale + alibi + key padding mask
#pragma unroll
    for (int sub = 0; sub < 2; ++sub) {
      const int col = j + sub * 16 + nc;
      const bool msk = pmask[b * kN + col] != 0;
      const float* al = alibi + ((size_t)h * kN + q0 + 8 * hi) * kN + col;
#pragma unroll
      for (int r = 0; r < 8; ++r) {
        float sc = msk ? -1e30f : (s[sub][r] * kScale + al[(size_t)r * kN]);
        s[sub][r] = sc;
      }
    }
    // per-row max over the chunk (rows are striped across 16-lane halves)
    float mnew[8];
#pragma unroll
    for (int r = 0; r < 8; ++r) {
      float vmx = fmaxf(s[0][r], s[1][r]);
#pragma unroll
      for (int off = 1; off < 16; off <<= 1)
        vmx = fmaxf(vmx, __shfl_xor(vmx, off, 32));
      mnew[r] = fmaxf(mrow[r], vmx);
    }
#pragma unroll
    for (int r = 0; r < 8; ++r) {
      const float f = __expf(mrow[r] - mnew[r]);
      mrow[r] = mnew[r];
      lrow[r] *= f;
      o[0][r] *= f; o[1][r] *= f; o[2][r] *= f; o[3][r] *= f;
    }
    // P = exp(S - m); stash in LDS in [row][col] order; accumulate row sums
    float ps[8] = {};
#pragma unroll
    for (int sub = 0; sub < 2; ++sub) {
#pragma unroll
      for (int r = 0; r < 8; ++r) {
        const float p = __expf(s[sub][r] - mrow[r]);
        myp[(r + 8 * hi) * 32 + sub * 16 + nc] = p;
        ps[r] += p;
      }
    }
#pragma unroll
    for (int r = 0; r < 8; ++r) {
      float t = ps[r];
#pragma unroll
      for (int off = 1; off < 16; off <<= 1) t += __shfl_xor(t, off, 32);
      lrow[r] += t;
    }
    __builtin_amdgcn_wave_barrier();
    asm volatile("s_wait_dscnt 0" ::: "memory");  // wave-local LDS transpose sync
    // Re-read P as a 16x32 bf16 A-fragment
    v16bf pa;
    {
      const int base = hi ? 8 : 0;
      const float* prow = myp + nc * 32;
#pragma unroll
      for (int i = 0; i < 8; ++i) {
        pa[i]     = f2bf(prow[base + i]);
        pa[8 + i] = f2bf(prow[base + 16 + i]);
      }
    }
    __builtin_amdgcn_wave_barrier();
    // O(16x64) += P(16x32) x V(32x64): issue all 4 V fragments, then 4 WMMAs
    v16bf vb0 = load_b_frag(vtp + (size_t)(0 * 16 + nc) * kN + j, lane);
    v16bf vb1 = load_b_frag(vtp + (size_t)(1 * 16 + nc) * kN + j, lane);
    v16bf vb2 = load_b_frag(vtp + (size_t)(2 * 16 + nc) * kN + j, lane);
    v16bf vb3 = load_b_frag(vtp + (size_t)(3 * 16 + nc) * kN + j, lane);
    o[0] = wmma_bf16(pa, vb0, o[0]);
    o[1] = wmma_bf16(pa, vb1, o[1]);
    o[2] = wmma_bf16(pa, vb2, o[2]);
    o[3] = wmma_bf16(pa, vb3, o[3]);
  }
  // normalize and store A in [B,H,N,D] bf16
#pragma unroll
  for (int d = 0; d < 4; ++d) {
#pragma unroll
    for (int r = 0; r < 8; ++r) {
      const int row = q0 + r + 8 * hi;
      Abf[((size_t)(b * kH + h) * kN + row) * kD + d * 16 + nc] =
          f2bf(o[d][r] / lrow[r]);
    }
  }
}

// ---- Gated linear fusion: C2/g from q and A, elementwise gate ------------

__global__ __launch_bounds__(128) void gated_fusion_kernel(
    const __bf16* __restrict__ qbf, const __bf16* __restrict__ Abf,
    const __bf16* __restrict__ WqCb, const __bf16* __restrict__ WACb,
    const __bf16* __restrict__ Wqgb, const __bf16* __restrict__ WAgb,
    const float* __restrict__ bC, const float* __restrict__ bg,
    __bf16* __restrict__ aega) {
  const int lane = threadIdx.x & 31;
  const int wv   = threadIdx.x >> 5;        // col tile 0..3 (D = 64)
  const int row0 = blockIdx.x * 16;         // 4096 row tiles over B*H*N
  const int col0 = wv * 16;
  const int nc = lane & 15, hi = (lane >> 4) & 1;
  const __bf16* qrow = qbf + (size_t)(row0 + nc) * kD;
  const __bf16* Arow = Abf + (size_t)(row0 + nc) * kD;
  const v16bf qa0 = load_a_frag(qrow, lane),      qa1 = load_a_frag(qrow + 32, lane);
  const v16bf Aa0 = load_a_frag(Arow, lane),      Aa1 = load_a_frag(Arow + 32, lane);
  const __bf16* wq = WqCb + (size_t)(col0 + nc) * kD;
  const __bf16* wa = WACb + (size_t)(col0 + nc) * kD;
  const __bf16* gq = Wqgb + (size_t)(col0 + nc) * kD;
  const __bf16* ga = WAgb + (size_t)(col0 + nc) * kD;
  v16bf wq0 = load_b_frag(wq, lane),      wq1 = load_b_frag(wq + 32, lane);
  v16bf wa0 = load_b_frag(wa, lane),      wa1 = load_b_frag(wa + 32, lane);
  v16bf gq0 = load_b_frag(gq, lane),      gq1 = load_b_frag(gq + 32, lane);
  v16bf ga0 = load_b_frag(ga, lane),      ga1 = load_b_frag(ga + 32, lane);
  v8f c2 = {};
  c2 = wmma_bf16(qa0, wq0, c2);
  c2 = wmma_bf16(qa1, wq1, c2);
  c2 = wmma_bf16(Aa0, wa0, c2);
  c2 = wmma_bf16(Aa1, wa1, c2);
  v8f gg = {};
  gg = wmma_bf16(qa0, gq0, gg);
  gg = wmma_bf16(qa1, gq1, gg);
  gg = wmma_bf16(Aa0, ga0, gg);
  gg = wmma_bf16(Aa1, ga1, gg);
  const int e = col0 + nc;
  const float bce = bC[e], bge = bg[e];
#pragma unroll
  for (int r = 0; r < 8; ++r) {
    const float C2v = c2[r] + bce;
    const float gv  = gg[r] + bge;
    const float sg  = 1.f / (1.f + __expf(-gv));
    const float G   = (__expf(sg) + 1.f) * kSuppress;
    const int m  = row0 + r + 8 * hi;       // (b, h, n) flat
    const int bb = m >> 14;
    const int hh = (m >> 10) & (kH - 1);
    const int n  = m & (kN - 1);
    aega[((size_t)(bb * kN + n) * kH + hh) * kD + e] = f2bf(G * C2v);
  }
}

// ---- Output projection: Aega[4096,1024] @ Wo^T + bo -> f32 out -----------
// Same 16x64-per-wave register blocking as the QKV GEMM.

__global__ __launch_bounds__(128) void out_proj_kernel(
    const __bf16* __restrict__ aega, const __bf16* __restrict__ Wob,
    const float* __restrict__ bo, float* __restrict__ out) {
  const int lane = threadIdx.x & 31;
  const int wv   = threadIdx.x >> 5;
  const int mt   = blockIdx.y * 4 + wv;    // 256 row tiles
  const int cb   = blockIdx.x;             // 16 col blocks of 64
  const int row0 = mt * 16, col0 = cb * 64;
  const int nc = lane & 15, hi = (lane >> 4) & 1;
  const __bf16* arow  = aega + (size_t)(row0 + nc) * kDIM;
  const __bf16* brow0 = Wob  + (size_t)(col0 + nc) * kDIM;
  v8f acc[4] = {{}, {}, {}, {}};
  for (int k = 0; k < kDIM; k += 32) {
    v16bf a  = load_a_frag(arow + k, lane);
    v16bf b0 = load_b_frag(brow0 + (size_t)0 * 16 * kDIM + k, lane);
    v16bf b1 = load_b_frag(brow0 + (size_t)1 * 16 * kDIM + k, lane);
    v16bf b2 = load_b_frag(brow0 + (size_t)2 * 16 * kDIM + k, lane);
    v16bf b3 = load_b_frag(brow0 + (size_t)3 * 16 * kDIM + k, lane);
    acc[0] = wmma_bf16(a, b0, acc[0]);
    acc[1] = wmma_bf16(a, b1, acc[1]);
    acc[2] = wmma_bf16(a, b2, acc[2]);
    acc[3] = wmma_bf16(a, b3, acc[3]);
  }
#pragma unroll
  for (int t = 0; t < 4; ++t) {
    const int col = col0 + t * 16 + nc;
    const float bsc = bo[col];
#pragma unroll
    for (int r = 0; r < 8; ++r) {
      const int m = row0 + r + 8 * hi;
      out[(size_t)m * kDIM + col] = acc[t][r] + bsc;
    }
  }
}

// ---- host launch ---------------------------------------------------------

extern "C" void kernel_launch(void* const* d_in, const int* in_sizes, int n_in,
                              void* d_out, int out_size, void* d_ws, size_t ws_size,
                              hipStream_t stream) {
  (void)in_sizes; (void)n_in; (void)out_size; (void)ws_size;
  const float* x      = (const float*)d_in[0];
  const unsigned char* pm = (const unsigned char*)d_in[1];
  const float* alibi  = (const float*)d_in[2];
  const float* Wq  = (const float*)d_in[3];  const float* bq = (const float*)d_in[4];
  const float* Wk  = (const float*)d_in[5];  const float* bk = (const float*)d_in[6];
  const float* Wv  = (const float*)d_in[7];  const float* bv = (const float*)d_in[8];
  const float* WqC = (const float*)d_in[9];  const float* WAC = (const float*)d_in[10];
  const float* bC  = (const float*)d_in[11];
  const float* Wqg = (const float*)d_in[12]; const float* WAg = (const float*)d_in[13];
  const float* bg  = (const float*)d_in[14];
  const float* Wo  = (const float*)d_in[15]; const float* bo = (const float*)d_in[16];

  char* ws = (char*)d_ws;
  size_t off = 0;
  auto carve = [&](size_t elems) {
    __bf16* p = (__bf16*)(ws + off);
    off = (off + elems * sizeof(__bf16) + 255) & ~(size_t)255;
    return p;
  };
  const size_t nTok = (size_t)kB * kN;                 // 4096 tokens
  __bf16* xbf   = carve(nTok * kDIM);
  __bf16* Wqb   = carve((size_t)kDIM * kDIM);
  __bf16* Wkb   = carve((size_t)kDIM * kDIM);
  __bf16* Wvb   = carve((size_t)kDIM * kDIM);
  __bf16* Wob   = carve((size_t)kDIM * kDIM);
  __bf16* WqCb  = carve((size_t)kD * kD);
  __bf16* WACb  = carve((size_t)kD * kD);
  __bf16* Wqgb  = carve((size_t)kD * kD);
  __bf16* WAgb  = carve((size_t)kD * kD);
  __bf16* qb    = carve(nTok * kDIM);   // [B,H,N,D]
  __bf16* kb2   = carve(nTok * kDIM);   // [B,H,N,D]
  __bf16* vtb   = carve(nTok * kDIM);   // [B,H,D,N]
  __bf16* Ab    = carve(nTok * kDIM);   // [B,H,N,D]
  __bf16* aegab = carve(nTok * kDIM);   // [B,N,H,D]

  auto cvt = [&](const float* s, __bf16* d, int n) {
    cvt_bf16_kernel<<<(n + 255) / 256, 256, 0, stream>>>(s, d, n);
  };
  cvt(x, xbf, (int)(nTok * kDIM));
  cvt(Wq, Wqb, kDIM * kDIM);
  cvt(Wk, Wkb, kDIM * kDIM);
  cvt(Wv, Wvb, kDIM * kDIM);
  cvt(Wo, Wob, kDIM * kDIM);
  cvt(WqC, WqCb, kD * kD);
  cvt(WAC, WACb, kD * kD);
  cvt(Wqg, Wqgb, kD * kD);
  cvt(WAg, WAgb, kD * kD);

  qkv_proj_kernel<<<dim3(16, 64, 3), 128, 0, stream>>>(
      xbf, Wqb, Wkb, Wvb, bq, bk, bv, qb, kb2, vtb);
  flash_attn_kernel<<<dim3(16, kH, kB), 128, 0, stream>>>(
      qb, kb2, vtb, alibi, pm, Ab);
  gated_fusion_kernel<<<4096, 128, 0, stream>>>(
      qb, Ab, WqCb, WACb, Wqgb, WAgb, bC, bg, aegab);
  out_proj_kernel<<<dim3(16, 64), 128, 0, stream>>>(
      aegab, Wob, bo, (float*)d_out);
}
